// Feedback_70093866271267
// MI455X (gfx1250) — matrix operand order repaired and checked by reading, not compiled, not run on hardware
//
#include <hip/hip_runtime.h>
#include <math.h>

typedef __attribute__((ext_vector_type(16))) _Float16 v16h;
typedef __attribute__((ext_vector_type(4)))  _Float16 v4h;
typedef __attribute__((ext_vector_type(8)))  float    v8f;
typedef int v4i __attribute__((vector_size(4 * sizeof(int))));

#define BM 128
#define BN 128
#define BK 32
#define LDS_K (BK + 8)   // 40-half rows (80B): keeps 16B alignment for b128 frag loads

#if defined(__has_builtin)
#if __has_builtin(__builtin_amdgcn_global_load_async_to_lds_b128) && \
    __has_builtin(__builtin_amdgcn_s_wait_asynccnt)
#define HAVE_ASYNC_LDS 1
#endif
#endif
#ifndef HAVE_ASYNC_LDS
#define HAVE_ASYNC_LDS 0
#endif

union Frag16 { uint4 u[2]; v16h h; };

#if HAVE_ASYNC_LDS
__device__ __forceinline__ void async_cp_b128(const _Float16* g, _Float16* l)
{
    __builtin_amdgcn_global_load_async_to_lds_b128(
        (__attribute__((address_space(1))) v4i*)const_cast<_Float16*>(g),
        (__attribute__((address_space(3))) v4i*)l, 0, 0);
}
#endif

// Generic GEMM on the f16 WMMA path:
//   C[m][n] = act( sum_k A(m,k) * W(k,n) + bias[n] )
// A side: fp32, converted to f16 while staging; optional K-split A1(K1)+A2(K2)
//   (LSTM step z = x@Wx + h@Wh in one pass; K1 % 32 == 0 so the select is a
//   uniform per-tile branch).
// B side: either WT (pre-packed f16 [N][Ktot], async global->LDS staging) or
//   W1 (fp32 [Ktot][N], strided convert-on-stage with next-tile prefetch).
// Double-buffered LDS: tile k+1 is staged (async DMA + global loads) while
// tile k computes; one barrier per k-tile.
// rowsPerBatch>0 remaps row m -> (b=m/rpb, r=m%rpb) for A1 and C (cand GEMM).
__global__ __launch_bounds__(256)
void wmma_gemm_f32(const float* __restrict__ A1, long lda1, int K1,
                   const float* __restrict__ A2, long lda2, int K2,
                   const float* __restrict__ W1, long ldw1,
                   const _Float16* __restrict__ WT, long ldwt,
                   const float* __restrict__ bias,
                   float* __restrict__ C, long ldc,
                   int M, int N,
                   int rowsPerBatch, long aBatchStride, long cBatchStride,
                   int act)
{
    __shared__ __align__(16) _Float16 sA[2][BM][LDS_K];
    __shared__ __align__(16) _Float16 sB[2][BN][LDS_K];

    const int tid  = threadIdx.x;
    const int lane = tid & 31;
    const int wave = tid >> 5;
    const int wm   = wave >> 1;        // 4 wave-rows of 32 M each
    const int wn   = wave & 1;         // 2 wave-cols of 64 N each
    const int blockM = blockIdx.y * BM;
    const int blockN = blockIdx.x * BN;
    const int l16 = lane & 15;
    const int lHi = lane >> 4;         // 0 or 1

    const int Ktot = K1 + K2;
    const int nK   = Ktot / BK;        // all problem K are multiples of 32

    // ---- staging helpers ----
    auto stageB = [&](int buf, int k0) {
        if (WT) {
            // pre-packed f16 [N][Ktot]: contiguous 64B per n-row, async DMA
            int n  = tid >> 1;
            int kh = (tid & 1) << 4;                 // 0 or 16 halfs
            const _Float16* srcW = WT + (long)(blockN + n) * ldwt + (k0 + kh);
#if HAVE_ASYNC_LDS
            async_cp_b128(srcW,     &sB[buf][n][kh]);
            async_cp_b128(srcW + 8, &sB[buf][n][kh + 8]);
#else
            *(uint4*)&sB[buf][n][kh]     = *(const uint4*)srcW;
            *(uint4*)&sB[buf][n][kh + 8] = *(const uint4*)(srcW + 8);
#endif
        } else {
            // fp32 [Ktot][N]: transpose + convert while staging
            int n  = tid & 127;
            int kh = tid >> 7;                       // 0..1
            int gn = blockN + n;
            const float* wp = W1 + (long)(k0 + kh * 16) * ldw1 + gn;
            float wv[16];
#pragma unroll
            for (int kk = 0; kk < 16; ++kk) { wv[kk] = *wp; wp += ldw1; }
#pragma unroll
            for (int kk = 0; kk < 16; ++kk)
                sB[buf][n][kh * 16 + kk] = (_Float16)wv[kk];
            int pk = k0 + BK + kh * 16;              // prefetch 1 tile ahead of staged
            if (pk < Ktot) __builtin_prefetch(W1 + (long)pk * ldw1 + gn, 0, 1);
        }
    };

    auto stageA = [&](int buf, int k0) {
        // K1 % 32 == 0, so the whole tile comes from exactly one of A1/A2.
        const bool useA2 = (k0 >= K1);
        float4 va[4];
#pragma unroll
        for (int it = 0; it < 4; ++it) {
            int idx = it * 256 + tid;
            int row = idx >> 3;
            int c4  = (idx & 7) << 2;
            int gm  = blockM + row;
            const float* src;
            if (useA2) {
                src = A2 + (long)gm * lda2 + (k0 - K1 + c4);
            } else if (rowsPerBatch > 0) {
                int bb = gm / rowsPerBatch;
                int rr = gm - bb * rowsPerBatch;
                src = A1 + (long)bb * aBatchStride + (long)rr * lda1 + (k0 + c4);
            } else {
                src = A1 + (long)gm * lda1 + (k0 + c4);
            }
            va[it] = *(const float4*)src;
        }
#pragma unroll
        for (int it = 0; it < 4; ++it) {
            int idx = it * 256 + tid;
            int row = idx >> 3;
            int c4  = (idx & 7) << 2;
            v4h h4;
            h4.x = (_Float16)va[it].x; h4.y = (_Float16)va[it].y;
            h4.z = (_Float16)va[it].z; h4.w = (_Float16)va[it].w;
            *(v4h*)&sA[buf][row][c4] = h4;
        }
    };

    v8f acc[2][4];
    const v8f vzero = {0.f,0.f,0.f,0.f,0.f,0.f,0.f,0.f};
#pragma unroll
    for (int i = 0; i < 2; ++i)
#pragma unroll
        for (int j = 0; j < 4; ++j) acc[i][j] = vzero;

    // ---- prologue: stage tile 0 ----
    stageB(0, 0);
    stageA(0, 0);
#if HAVE_ASYNC_LDS
    if (WT) __builtin_amdgcn_s_wait_asynccnt(0);
#endif
    __syncthreads();

    // ---- pipelined main loop: one barrier per k-tile ----
    for (int kt = 0; kt < nK; ++kt) {
        const int cur = kt & 1;
        const bool hasNext = (kt + 1) < nK;
        if (hasNext) {                 // stage tile kt+1 into the other buffer
            stageB(cur ^ 1, (kt + 1) * BK);
            stageA(cur ^ 1, (kt + 1) * BK);
        }

        // fragments of tile kt (ISA 16-bit A 16x32 / B 32x16 lane layouts)
        Frag16 aF[2], bF[4];
#pragma unroll
        for (int mi = 0; mi < 2; ++mi) {
            int row = wm * 32 + mi * 16 + l16;
            int kc  = lHi ? 8 : 0;     // lanes 0-15: K{0..7,16..23}; 16-31: K{8..15,24..31}
            aF[mi].u[0] = *(const uint4*)&sA[cur][row][kc];
            aF[mi].u[1] = *(const uint4*)&sA[cur][row][kc + 16];
        }
#pragma unroll
        for (int ni = 0; ni < 4; ++ni) {
            int col = wn * 64 + ni * 16 + l16;
            int kc  = lHi ? 8 : 0;     // lanes 0-15: K{0..7,16..23}; 16-31: K{8..15,24..31}
            bF[ni].u[0] = *(const uint4*)&sB[cur][col][kc];
            bF[ni].u[1] = *(const uint4*)&sB[cur][col][kc + 16];
        }

#pragma unroll
        for (int mi = 0; mi < 2; ++mi)
#pragma unroll
            for (int ni = 0; ni < 4; ++ni)
                acc[mi][ni] = __builtin_amdgcn_wmma_f32_16x16x32_f16(
                    false, aF[mi].h, false, bF[ni].h, (short)0, acc[mi][ni],
                    false, false);

        if (hasNext) {
#if HAVE_ASYNC_LDS
            if (WT) __builtin_amdgcn_s_wait_asynccnt(0);
#endif
            __syncthreads();
        }
    }

    // ---- epilogue: bias + optional leaky, fp32 store ----
#pragma unroll
    for (int mi = 0; mi < 2; ++mi) {
        int rowBase = blockM + wm * 32 + mi * 16 + lHi * 8;
#pragma unroll
        for (int ni = 0; ni < 4; ++ni) {
            int col = blockN + wn * 64 + ni * 16 + l16;
            float bv = bias ? bias[col] : 0.0f;
#pragma unroll
            for (int r = 0; r < 8; ++r) {
                int m = rowBase + r;
                float v = acc[mi][ni][r] + bv;
                if (act == 1) v = (v >= 0.0f) ? v : 0.01f * v;
                long off;
                if (rowsPerBatch > 0) {
                    int bb = m / rowsPerBatch;
                    int rr = m - bb * rowsPerBatch;
                    off = (long)bb * cBatchStride + (long)rr * ldc + col;
                } else {
                    off = (long)m * ldc + col;
                }
                C[off] = v;
            }
        }
    }
}

// Pack fp32 W1[K1,N] (+ optional W2[K2,N]) into f16 WT[N][K1+K2] (one-time).
__global__ void pack_wT_kernel(const float* __restrict__ W1,
                               const float* __restrict__ W2,
                               int K1, int Ktot, int N,
                               _Float16* __restrict__ WT)
{
    int k = blockIdx.x * 256 + threadIdx.x;   // grid sized exactly
    int n = blockIdx.y;
    float w = (k < K1) ? W1[(long)k * N + n] : W2[(long)(k - K1) * N + n];
    WT[(long)n * Ktot + k] = (_Float16)w;
}

__device__ __forceinline__ float sigf(float x) { return 1.0f / (1.0f + __expf(-x)); }

// gates i,f,g,o at columns j, j+1024, j+2048, j+3072 of z
__global__ void lstm_gates_kernel(const float* __restrict__ z,
                                  float* __restrict__ c, float* __restrict__ h)
{
    int idx = blockIdx.x * blockDim.x + threadIdx.x;   // B*H = 1<<20
    int b = idx >> 10;
    int j = idx & 1023;
    const float* zr = z + (long)b * 4096;
    float gi = sigf(zr[j]);
    float gf = sigf(zr[j + 1024]);
    float gg = tanhf(zr[j + 2048]);
    float go = sigf(zr[j + 3072]);
    float cc = gf * c[idx] + gi * gg;
    c[idx] = cc;
    h[idx] = go * tanhf(cc);
}

__global__ void store_pred_kernel(const float* __restrict__ pred,
                                  float* __restrict__ outPred, int t)
{
    int idx = blockIdx.x * blockDim.x + threadIdx.x;   // B*F = 262144
    int b = idx >> 8;
    int f = idx & 255;
    outPred[((long)b * 96 + t) * 256 + f] = pred[idx];
}

__global__ void copy_h_kernel(const float* __restrict__ h, float* __restrict__ catin)
{
    int idx = blockIdx.x * blockDim.x + threadIdx.x;   // B*H = 1<<20
    int b = idx >> 10;
    int j = idx & 1023;
    catin[(long)b * 32768 + j] = h[idx];
}

// 2x2 W_pf + softmax over candidate pairs
__global__ void prob_head_kernel(const float* __restrict__ prob,
                                 const float* __restrict__ Wpf,
                                 const float* __restrict__ bpf,
                                 float* __restrict__ outProb)
{
    int idx = blockIdx.x * blockDim.x + threadIdx.x;   // B*Tm*ns*mc = 12,582,912
    int b   = idx / 12288;
    int rem = idx - b * 12288;
    const float* p = prob + (long)b * 24576 + (long)rem * 2;
    float p0 = p[0], p1 = p[1];
    float y0 = p0 * Wpf[0] + p1 * Wpf[2] + bpf[0];
    float y1 = p0 * Wpf[1] + p1 * Wpf[3] + bpf[1];
    float mx = fmaxf(y0, y1);
    float e0 = __expf(y0 - mx), e1 = __expf(y1 - mx);
    float inv = 1.0f / (e0 + e1);
    outProb[(long)idx * 2]     = e0 * inv;
    outProb[(long)idx * 2 + 1] = e1 * inv;
}

extern "C" void kernel_launch(void* const* d_in, const int* in_sizes, int n_in,
                              void* d_out, int out_size, void* d_ws, size_t ws_size,
                              hipStream_t stream)
{
    const float* inputs = (const float*)d_in[0];
    const float* Wx     = (const float*)d_in[1];
    const float* Wh     = (const float*)d_in[2];
    const float* b_lstm = (const float*)d_in[3];
    const float* W_warm = (const float*)d_in[4];
    const float* b_warm = (const float*)d_in[5];
    const float* W_cand = (const float*)d_in[6];
    const float* b_cand = (const float*)d_in[7];
    const float* W_cat  = (const float*)d_in[8];
    const float* b_cat  = (const float*)d_in[9];
    const float* W_cost = (const float*)d_in[10];
    const float* b_cost = (const float*)d_in[11];
    const float* W_prob = (const float*)d_in[12];
    const float* b_prob = (const float*)d_in[13];
    const float* W_pf   = (const float*)d_in[14];
    const float* b_pf   = (const float*)d_in[15];
    (void)in_sizes; (void)n_in; (void)out_size; (void)ws_size;

    char* ws = (char*)d_ws;                          // layout (~268 MB total)
    float*     z      = (float*)(ws);                // 1024 x 4096         (16 MB)
    float*     hbuf   = (float*)(ws + (16l  << 20)); // 1024 x 1024         ( 4 MB)
    float*     cbuf   = (float*)(ws + (20l  << 20)); // 1024 x 1024         ( 4 MB)
    float*     pred   = (float*)(ws + (24l  << 20)); // 1024 x 256          ( 1 MB)
    float*     catin  = (float*)(ws + (25l  << 20)); // 1024 x 32768        (128 MB)
    float*     cat    = (float*)(ws + (153l << 20)); // 1024 x 2048         ( 8 MB)
    float*     prob   = (float*)(ws + (161l << 20)); // 1024 x 24576        (96 MB)
    _Float16*  WzT    = (_Float16*)(ws + (257l << 20)); // [4096][1280] f16 (10 MB)
    _Float16*  WwarmT = (_Float16*)(ws + (267l << 20)); // [256][1024] f16  (0.5 MB)

    float* outPred = (float*)d_out;              // [B,Tm,ns,mc,2] = 25,165,824
    float* outCost = outPred + 25165824l;        // [B,ns,mc]      =    131,072
    float* outProb = outCost + 131072l;          // [B,Tm,ns,mc,2] = 25,165,824
    float* outInp  = outProb + 25165824l;        // [B,T,ns,mc,2]  = 33,554,432

    (void)hipMemsetAsync(hbuf, 0, (size_t)1024 * 1024 * 4, stream);
    (void)hipMemsetAsync(cbuf, 0, (size_t)1024 * 1024 * 4, stream);

    // one-time packs: WzT = [Wx;Wh] -> f16 [4096][1280]; WwarmT -> f16 [256][1024]
    pack_wT_kernel<<<dim3(5, 4096), 256, 0, stream>>>(Wx, Wh, 256, 1280, 4096, WzT);
    pack_wT_kernel<<<dim3(4, 256),  256, 0, stream>>>(W_warm, nullptr, 1024, 1024, 256, WwarmT);

    dim3 blk(256);
    auto gemm = [&](const float* A1, long lda1, int K1,
                    const float* A2, long lda2, int K2,
                    const float* W1, long ldw1,
                    const _Float16* WT, long ldwt,
                    const float* bias, float* Cp, long ldc, int M, int N,
                    int rpb, long abs_, long cbs, int act) {
        dim3 grid(N / BN, M / BM);
        hipLaunchKernelGGL(wmma_gemm_f32, grid, blk, 0, stream,
                           A1, lda1, K1, A2, lda2, K2, W1, ldw1, WT, ldwt,
                           bias, Cp, ldc, M, N, rpb, abs_, cbs, act);
    };

    // ---- warm-up: 96 LSTM steps over inputs[:, 32:] ----
    for (int t = 0; t < 96; ++t) {
        gemm(inputs + (long)(32 + t) * 256, 32768, 256,   // x_t, lda = T*F
             hbuf, 1024, 1024,                            // h
             nullptr, 0, WzT, 1280, b_lstm,
             z, 4096, 1024, 4096, 0, 0, 0, 0);
        lstm_gates_kernel<<<4096, 256, 0, stream>>>(z, cbuf, hbuf);
    }

    // snapshot warm-up h into cat input columns [0,1024)
    copy_h_kernel<<<4096, 256, 0, stream>>>(hbuf, catin);

    // pred0 = h @ W_warm + b_warm
    gemm(hbuf, 1024, 1024, nullptr, 0, 0, nullptr, 0, WwarmT, 1024,
         b_warm, pred, 256, 1024, 256, 0, 0, 0, 0);
    store_pred_kernel<<<1024, 256, 0, stream>>>(pred, outPred, 0);

    // ---- autoregressive feedback: 95 steps, x = previous pred ----
    for (int s = 1; s < 96; ++s) {
        gemm(pred, 256, 256, hbuf, 1024, 1024,
             nullptr, 0, WzT, 1280, b_lstm, z, 4096, 1024, 4096, 0, 0, 0, 0);
        lstm_gates_kernel<<<4096, 256, 0, stream>>>(z, cbuf, hbuf);
        gemm(hbuf, 1024, 1024, nullptr, 0, 0, nullptr, 0, WwarmT, 1024,
             b_warm, pred, 256, 1024, 256, 0, 0, 0, 0);
        store_pred_kernel<<<1024, 256, 0, stream>>>(pred, outPred, s);
    }

    // ---- cand = leaky(inputs[:, :31] @ W_cand + b_cand) -> catin[:, 1024:] ----
    gemm(inputs, 256, 256, nullptr, 0, 0, W_cand, 1024, nullptr, 0, b_cand,
         catin + 1024, 1024, 31744, 1024,
         /*rowsPerBatch=*/31, /*aBatchStride=*/32768, /*cBatchStride=*/32768, 1);

    // ---- cat = leaky(catin @ W_cat + b_cat) ----
    gemm(catin, 32768, 32768, nullptr, 0, 0, W_cat, 2048, nullptr, 0,
         b_cat, cat, 2048, 1024, 2048, 0, 0, 0, 1);

    // ---- cost -> d_out directly ----
    gemm(cat, 2048, 2048, nullptr, 0, 0, W_cost, 128, nullptr, 0,
         b_cost, outCost, 128, 1024, 128, 0, 0, 0, 0);

    // ---- prob, then fused 2x2 pf + softmax ----
    gemm(cat, 2048, 2048, nullptr, 0, 0, W_prob, 24576, nullptr, 0,
         b_prob, prob, 24576, 1024, 24576, 0, 0, 0, 0);
    prob_head_kernel<<<49152, 256, 0, stream>>>(prob, W_pf, b_pf, outProb);

    // ---- inp_r is just inputs reshaped ----
    (void)hipMemcpyAsync(outInp, inputs, (size_t)33554432 * 4,
                         hipMemcpyDeviceToDevice, stream);
}
